// BinaryTreeLSTM_29394756173864
// MI455X (gfx1250) — matrix operand order repaired and checked by reading, not compiled, not run on hardware
//
#include <hip/hip_runtime.h>
#include <hip/hip_bf16.h>
#include <math.h>

// ---------------------------------------------------------------------------
// BinaryTreeLSTM on MI455X (gfx1250, wave32, WMMA).
//
// Only the first H=128 columns of each gate survive each level, so we use the
// reduced weight set (512 rows). GEMM per level tile:
//   G[32 nodes x 512] = X[32x128] @ WihT + H0[32x256] @ WhhT  (+ bias)
// done with v_wmma_f32_16x16x32_f16. Wave w owns gate units [16w,16w+16) for
// ALL four gates, so i/f/g/o for a unit land in one wave's registers and the
// LSTM pointwise math needs no cross-wave traffic.
// ---------------------------------------------------------------------------

typedef __attribute__((ext_vector_type(16))) _Float16 v16h;
typedef __attribute__((ext_vector_type(8)))  _Float16 v8h;
typedef __attribute__((ext_vector_type(8)))  float    v8f;

#define H      128
#define NGATE  512          // used gate outputs (4 gates x 128 units)
#define LDSX_S 136          // 128 + 8 halfword pad (bank rotation)
#define LDSH_S 264          // 256 + 8 halfword pad

__device__ __forceinline__ v16h cat8(v8h a, v8h b) {
    v16h r;
#pragma unroll
    for (int i = 0; i < 8; ++i) { r[i] = a[i]; r[8 + i] = b[i]; }
    return r;
}

// A fragment (16x32 f16, ISA layout): lane L holds row m=L&15,
// K = {k0+8*half..+7} U {k0+16+8*half..+7}, half = L>>4.
__device__ __forceinline__ v16h load_a_lds(const _Float16* s, int stride,
                                           int m, int k0, int half) {
    const _Float16* p = s + m * stride + k0 + half * 8;
    v8h a0 = *(const v8h*)(p);
    v8h a1 = *(const v8h*)(p + 16);
    return cat8(a0, a1);
}

// B fragment (32x16 f16): lane L holds column j=base+(L&15),
// K = {k0+8*half..+7} U {k0+16+8*half..+7}, half = L>>4.
__device__ __forceinline__ v16h load_b_glb(const _Float16* __restrict__ W,
                                           int stride, int j, int k0, int half) {
    const _Float16* p = W + (size_t)j * stride + k0 + half * 8;
    v8h b0 = *(const v8h*)(p);
    v8h b1 = *(const v8h*)(p + 16);
    return cat8(b0, b1);
}

__device__ __forceinline__ v8f wmma_f16(v16h a, v16h b, v8f c) {
    return __builtin_amdgcn_wmma_f32_16x16x32_f16(
        false, a, false, b, (short)0, c, false, false);
}

__device__ __forceinline__ float sigm(float x) {
    return 1.0f / (1.0f + __expf(-x));
}

// Pack the 512 used weight rows (row g*256+u of the original 1024) to f16 and
// fuse the two biases.  j = g*128+u indexes the packed row.
__global__ void tlstm_prep(const float* __restrict__ W_ih,
                           const float* __restrict__ W_hh,
                           const float* __restrict__ b_ih,
                           const float* __restrict__ b_hh,
                           _Float16* __restrict__ Wih_h,
                           _Float16* __restrict__ Whh_h,
                           float* __restrict__ bias) {
    int j = blockIdx.x;              // 0..511
    int g = j >> 7, u = j & 127;
    int r = g * 256 + u;             // original row
    for (int k = threadIdx.x; k < 128; k += blockDim.x)
        Wih_h[(size_t)j * 128 + k] = (_Float16)W_ih[(size_t)r * 128 + k];
    for (int k = threadIdx.x; k < 256; k += blockDim.x)
        Whh_h[(size_t)j * 256 + k] = (_Float16)W_hh[(size_t)r * 256 + k];
    if (threadIdx.x == 0) bias[j] = b_ih[r] + b_hh[r];
}

// One tree level: n nodes, children h/c from previous level (nullptr => 0).
__global__ void __launch_bounds__(256)
tlstm_level(const float* __restrict__ emb,       // n x 128 (level slice)
            const _Float16* __restrict__ h_in,   // 2n x 128 f16 (or null)
            const float* __restrict__ c_in,      // 2n x 128 f32 (or null)
            _Float16* __restrict__ h_out,        // n x 128 f16
            float* __restrict__ c_out,           // n x 128 f32
            const _Float16* __restrict__ Wih,    // 512 x 128 f16
            const _Float16* __restrict__ Whh,    // 512 x 256 f16
            const float* __restrict__ bias,      // 512
            int n) {
    __shared__ __align__(16) _Float16 xs[32 * LDSX_S];   // x tile, f16
    __shared__ __align__(16) _Float16 hs[32 * LDSH_S];   // h0 tile, f16

    const int tid  = threadIdx.x;
    const int wave = tid >> 5;
    const int lane = tid & 31;
    const int half = lane >> 4;
    const int nb   = blockIdx.x * 32;            // first node of this block

    // ---- stage X (f32 -> f16) and H0 (already f16) into padded LDS --------
    for (int idx = tid; idx < 32 * 128; idx += 256) {
        int m = idx >> 7, k = idx & 127;
        int row = nb + m; if (row >= n) row = n - 1;
        xs[m * LDSX_S + k] = (_Float16)emb[(size_t)row * 128 + k];
    }
    if (h_in) {
        for (int idx = tid; idx < 32 * 256; idx += 256) {
            int m = idx >> 8, k = idx & 255;
            int cr = 2 * (nb + m) + (k >> 7);    // child row (left/right)
            if (cr >= 2 * n) cr = 2 * n - 1;
            hs[m * LDSH_S + k] = h_in[(size_t)cr * 128 + (k & 127)];
        }
    }
    __syncthreads();

    // ---- WMMA main loop ----------------------------------------------------
    // Wave w -> gate units u in [16w, 16w+16); packed W row j = t*128 + u.
    const int ucol = (wave << 4) + (lane & 15);

    const v8f zero = {0.f, 0.f, 0.f, 0.f, 0.f, 0.f, 0.f, 0.f};
    v8f acc[2][4];
#pragma unroll
    for (int mt = 0; mt < 2; ++mt)
#pragma unroll
        for (int t = 0; t < 4; ++t) acc[mt][t] = zero;

    // X part: K = 128 -> 4 chunks of 32
#pragma unroll
    for (int kc = 0; kc < 4; ++kc) {
        const int k0 = kc << 5;
        v16h a0 = load_a_lds(xs,                 LDSX_S, lane & 15, k0, half);
        v16h a1 = load_a_lds(xs + 16 * LDSX_S,   LDSX_S, lane & 15, k0, half);
#pragma unroll
        for (int t = 0; t < 4; ++t) {
            v16h b = load_b_glb(Wih, 128, t * 128 + ucol, k0, half);
            acc[0][t] = wmma_f16(a0, b, acc[0][t]);
            acc[1][t] = wmma_f16(a1, b, acc[1][t]);
        }
    }
    // H0 part: K = 256 -> 8 chunks of 32 (skipped at leaf level: h0 == 0)
    if (h_in) {
#pragma unroll
        for (int kc = 0; kc < 8; ++kc) {
            const int k0 = kc << 5;
            v16h a0 = load_a_lds(hs,               LDSH_S, lane & 15, k0, half);
            v16h a1 = load_a_lds(hs + 16 * LDSH_S, LDSH_S, lane & 15, k0, half);
#pragma unroll
            for (int t = 0; t < 4; ++t) {
                v16h b = load_b_glb(Whh, 256, t * 128 + ucol, k0, half);
                acc[0][t] = wmma_f16(a0, b, acc[0][t]);
                acc[1][t] = wmma_f16(a1, b, acc[1][t]);
            }
        }
    }

    // ---- epilogue: LSTM pointwise, all gates already in this wave ----------
    const float bi = bias[ucol];
    const float bf = bias[128 + ucol];
    const float bg = bias[256 + ucol];
    const float bo = bias[384 + ucol];

#pragma unroll
    for (int mt = 0; mt < 2; ++mt) {
#pragma unroll
        for (int r = 0; r < 8; ++r) {
            // C/D layout: VGPR r -> row (half*8 + r) of the 16-row tile
            const int mrow = mt * 16 + (half << 3) + r;
            const int node = nb + mrow;
            const bool ok  = (node < n);

            float iv = acc[mt][0][r] + bi;
            float fv = acc[mt][1][r] + bf;
            float gv = acc[mt][2][r] + bg;
            float ov = acc[mt][3][r] + bo;

            float c0 = 0.0f;
            if (c_in && ok) c0 = c_in[(size_t)(2 * node) * 128 + ucol]; // left child

            float cc = sigm(fv) * c0 + sigm(iv) * tanhf(gv);
            float hh = sigm(ov) * tanhf(cc);
            if (ok) {
                h_out[(size_t)node * 128 + ucol] = (_Float16)hh;
                c_out[(size_t)node * 128 + ucol] = cc;
            }
        }
    }
}

// Root output: concat(h, c) -> 256 floats.
__global__ void tlstm_gather(const _Float16* __restrict__ h,
                             const float* __restrict__ c,
                             float* __restrict__ out) {
    int t = threadIdx.x;
    if (t < 128) out[t] = (float)h[t];
    else         out[t] = c[t - 128];
}

extern "C" void kernel_launch(void* const* d_in, const int* in_sizes, int n_in,
                              void* d_out, int out_size, void* d_ws, size_t ws_size,
                              hipStream_t stream) {
    (void)in_sizes; (void)n_in; (void)out_size; (void)ws_size;

    const float* emb  = (const float*)d_in[0];   // (2^18-1) x 128
    const float* W_ih = (const float*)d_in[1];   // 1024 x 128
    const float* W_hh = (const float*)d_in[2];   // 1024 x 256
    const float* b_ih = (const float*)d_in[3];   // 1024
    const float* b_hh = (const float*)d_in[4];   // 1024

    char* ws = (char*)d_ws;
    // Workspace layout (needs ~192 MB + 1 MB):
    _Float16* Wih_h = (_Float16*)(ws);                       // 512*128*2 = 128 KB
    _Float16* Whh_h = (_Float16*)(ws + (512 * 128 * 2));     // 512*256*2 = 256 KB
    float*    bias  = (float*)   (ws + (512 * 128 * 2) + (512 * 256 * 2)); // 2 KB

    const size_t ROWS   = (size_t)1 << 17;       // max rows per ping-pong buffer
    const size_t hbytes = ROWS * 128 * sizeof(_Float16);
    const size_t cbytes = ROWS * 128 * sizeof(float);
    char* act = ws + (1 << 20);                  // 1 MB-aligned activation region
    _Float16* hbuf[2] = { (_Float16*)(act),
                          (_Float16*)(act + hbytes) };
    float*    cbuf[2] = { (float*)(act + 2 * hbytes),
                          (float*)(act + 2 * hbytes + cbytes) };

    tlstm_prep<<<dim3(512), dim3(128), 0, stream>>>(
        W_ih, W_hh, b_ih, b_hh, Wih_h, Whh_h, bias);

    for (int d = 17; d >= 0; --d) {
        const int nn = 1 << d;
        const int oi = d & 1;           // output buffer parity
        const int ii = oi ^ 1;          // input = previous level's output
        const float* emb_d = emb + (size_t)(nn - 1) * 128;
        const bool leaf = (d == 17);
        tlstm_level<<<dim3((nn + 31) / 32), dim3(256), 0, stream>>>(
            emb_d,
            leaf ? nullptr : hbuf[ii],
            leaf ? nullptr : cbuf[ii],
            hbuf[oi], cbuf[oi],
            Wih_h, Whh_h, bias, nn);
    }

    // d = 0 wrote parity 0; emit concat(h, c).
    tlstm_gather<<<dim3(1), dim3(256), 0, stream>>>(hbuf[0], cbuf[0], (float*)d_out);
}
